// CMA_52956946760163
// MI455X (gfx1250) — matrix operand order repaired and checked by reading, not compiled, not run on hardware
//
#include <hip/hip_runtime.h>
#include <hip/hip_bf16.h>

#define Cc   2000
#define Dd   2048
#define NS   20000
#define HIGH_TH 0.55f
#define MID_TH  0.25f
#define OT_REG  0.07f
#define OT_ITERS 30
// 2*log(2000+1e-12)
#define TWO_LOG_C 15.201804919084164f

static constexpr size_t C2 = (size_t)Cc * Cc;

typedef __attribute__((ext_vector_type(16))) __bf16        v16bf;
typedef __attribute__((ext_vector_type(8)))  float         v8f;
typedef __attribute__((ext_vector_type(4)))  unsigned int  uint4v;

union Frag { uint4v q[2]; v16bf v; };

// ---------------- block reduction helpers (wave32-safe, LDS based) ----------
__device__ __forceinline__ float blockSum(float v, float* sm) {
    int t = threadIdx.x; sm[t] = v; __syncthreads();
    for (int s = blockDim.x >> 1; s > 0; s >>= 1) {
        if (t < s) sm[t] += sm[t + s];
        __syncthreads();
    }
    float r = sm[0]; __syncthreads(); return r;
}
__device__ __forceinline__ float blockMax(float v, float* sm) {
    int t = threadIdx.x; sm[t] = v; __syncthreads();
    for (int s = blockDim.x >> 1; s > 0; s >>= 1) {
        if (t < s) sm[t] = fmaxf(sm[t], sm[t + s]);
        __syncthreads();
    }
    float r = sm[0]; __syncthreads(); return r;
}
__device__ __forceinline__ void blockArgMax(float v, int i, float* smv, int* smi,
                                            float& ov, int& oi) {
    int t = threadIdx.x; smv[t] = v; smi[t] = i; __syncthreads();
    for (int s = blockDim.x >> 1; s > 0; s >>= 1) {
        if (t < s) {
            float v2 = smv[t + s]; int i2 = smi[t + s];
            if (v2 > smv[t] || (v2 == smv[t] && i2 < smi[t])) { smv[t] = v2; smi[t] = i2; }
        }
        __syncthreads();
    }
    ov = smv[0]; oi = smi[0]; __syncthreads();
}

__device__ __forceinline__ unsigned short f2bf_rne(float x) {
    unsigned u = __float_as_uint(x);
    unsigned r = u + 0x7fffu + ((u >> 16) & 1u);
    return (unsigned short)(r >> 16);
}

// ---------------- 1) fused softmax + per-class accumulation ----------------
__global__ void cma_sm_stats(const float* __restrict__ logits,
                             const int* __restrict__ ids,
                             float* __restrict__ proto, float* __restrict__ counts) {
    __shared__ float srow[Cc];
    __shared__ float sm[256];
    int s = blockIdx.x;
    const float* row = logits + (size_t)s * Cc;
    for (int j = threadIdx.x; j < Cc; j += blockDim.x) srow[j] = row[j];
    __syncthreads();
    float m = -3.4e38f;
    for (int j = threadIdx.x; j < Cc; j += blockDim.x) m = fmaxf(m, srow[j]);
    m = blockMax(m, sm);
    float sum = 0.f;
    for (int j = threadIdx.x; j < Cc; j += blockDim.x) sum += __expf(srow[j] - m);
    sum = blockSum(sum, sm);
    float inv = 1.f / sum;
    int lbl = ids[s];
    float* prow = proto + (size_t)lbl * Cc;
    for (int j = threadIdx.x; j < Cc; j += blockDim.x)
        atomicAdd(prow + j, __expf(srow[j] - m) * inv);
    if (threadIdx.x == 0) atomicAdd(counts + lbl, 1.f);
}

// ---------------- 2) proto mean + entropy + active flags -------------------
__global__ void cma_proto_fin(float* __restrict__ proto, const float* __restrict__ counts,
                              float* __restrict__ ent, float* __restrict__ act) {
    __shared__ float sm[256];
    int c = blockIdx.x;
    float cnt = counts[c];
    float inv = 1.f / fmaxf(cnt, 1.f);
    float* row = proto + (size_t)c * Cc;
    float e = 0.f;
    for (int j = threadIdx.x; j < Cc; j += blockDim.x) {
        float p = row[j] * inv;
        row[j] = p;
        p = fmaxf(p, 1e-12f);
        e += p * __logf(p);
    }
    e = blockSum(e, sm);
    if (threadIdx.x == 0) { ent[c] = -e; act[c] = (cnt > 0.f) ? 1.f : 0.f; }
}

// ---------------- 3) row-normalize memory, convert to bf16 -----------------
__global__ void cma_norm_bf16(const float* __restrict__ mem, unsigned short* __restrict__ out) {
    __shared__ float sm[256];
    int c = blockIdx.x;
    const float* row = mem + (size_t)c * Dd;
    float ss = 0.f;
    for (int j = threadIdx.x; j < Dd; j += blockDim.x) { float x = row[j]; ss += x * x; }
    ss = blockSum(ss, sm);
    float inv = 1.f / fmaxf(sqrtf(ss), 1e-12f);
    for (int j = threadIdx.x; j < Dd; j += blockDim.x)
        out[(size_t)c * Dd + j] = f2bf_rne(row[j] * inv);
}

// ---------------- 4) feat = clip((A @ B^T + 1)/2) via bf16 WMMA ------------
// Guard-free tiling: 125 m-tiles = 25 blocks * 5 waves; 125 n-tiles = 25
// blocks * 5 tiles-per-wave. Every wave's work is always in range -> no
// divergent branches, EXEC stays all-ones around every WMMA.
__global__ void __launch_bounds__(160) cma_feat_wmma(const unsigned short* __restrict__ A,
                                                     const unsigned short* __restrict__ B,
                                                     float* __restrict__ feat) {
    const int lane = threadIdx.x & 31;
    const int wid  = threadIdx.x >> 5;          // 0..4
    const int mt   = blockIdx.y * 5 + wid;      // m tile, [0,125)
    const int n0   = blockIdx.x * 80;           // base column, [0,2000) step 80
    const int hi   = lane >> 4;
    const int l16  = lane & 15;

    // A fragment: lanes 0-15: K 0..15, lanes 16-31: K 16..31 (contiguous, matches B).
    const unsigned short* arow = A + (size_t)(mt * 16 + l16) * Dd + hi * 16;
    // B fragment: column n of B == row n of ir memory; lanes 0-15: K 0..15,
    // lanes 16-31: K 16..31 (16 contiguous halves per lane group).
    const unsigned short* brow = B + (size_t)(n0 + l16) * Dd + hi * 16;

    v8f acc[5] = {};
    for (int k0 = 0; k0 < Dd; k0 += 32) {
        Frag fa;
        const uint4v* pa = (const uint4v*)(arow + k0);
        fa.q[0] = pa[0];
        fa.q[1] = pa[1];
        Frag fb[5];
#pragma unroll
        for (int t = 0; t < 5; t++) {
            const uint4v* pb = (const uint4v*)(brow + (size_t)t * 16 * Dd + k0);
            fb[t].q[0] = pb[0];
            fb[t].q[1] = pb[1];
        }
#pragma unroll
        for (int t = 0; t < 5; t++)
            acc[t] = __builtin_amdgcn_wmma_f32_16x16x32_bf16(
                false, fa.v, false, fb[t].v, (short)0, acc[t], false, false);
    }
#pragma unroll
    for (int t = 0; t < 5; t++) {
        int col = n0 + t * 16 + l16;
#pragma unroll
        for (int r = 0; r < 8; r++) {
            int rowi = mt * 16 + r + hi * 8;
            float v = (acc[t][r] + 1.f) * 0.5f;
            v = fminf(fmaxf(v, 0.f), 1.f);
            feat[(size_t)rowi * Cc + col] = v;
        }
    }
}

// ---------------- 5) LDS-tiled transpose (coalesced both directions) -------
__global__ void cma_transpose(const float* __restrict__ in, float* __restrict__ outM) {
    __shared__ float tile[32][33];
    int bx = blockIdx.x * 32, by = blockIdx.y * 32;
    int tx = threadIdx.x & 31, ty = threadIdx.x >> 5;   // 32x8 threads
#pragma unroll
    for (int r = ty; r < 32; r += 8) {
        int gi = by + r, gj = bx + tx;
        tile[r][tx] = (gi < Cc && gj < Cc) ? in[(size_t)gi * Cc + gj] : 0.f;
    }
    __syncthreads();
#pragma unroll
    for (int r = ty; r < 32; r += 8) {
        int gi = bx + r, gj = by + tx;
        if (gi < Cc && gj < Cc) outM[(size_t)gi * Cc + gj] = tile[tx][r];
    }
}

// ---------------- 6) cost -> Sinkhorn kernel matrix K ----------------------
__global__ void cma_cost_exp(const float* __restrict__ pv, const float* __restrict__ piT,
                             const float* __restrict__ feat,
                             const float* __restrict__ entv, const float* __restrict__ enti,
                             const float* __restrict__ actv, const float* __restrict__ acti,
                             float* __restrict__ K) {
    size_t idx = (size_t)blockIdx.x * blockDim.x + threadIdx.x;
    if (idx >= C2) return;
    int i = (int)(idx / Cc), j = (int)(idx % Cc);
    float val = 0.f;
    if (actv[i] > 0.f && acti[j] > 0.f) {
        float pred = 0.5f * (pv[idx] + piT[idx]);   // piT is ir_proto^T, coalesced
        float conf = 1.f - (entv[i] + enti[j]) / TWO_LOG_C;
        conf = fminf(fmaxf(conf, 0.f), 1.f);
        float cost = 0.45f * (1.f - pred) + 0.40f * (1.f - feat[idx]) + 0.15f * (1.f - conf);
        val = __expf(-cost / OT_REG);              // masked entries stay 0 == exp * mask
    }
    K[idx] = val;
}

// ---------------- 7) marginals a, b and u=v=1 ------------------------------
__global__ void cma_prep_ab(const float* __restrict__ cv, const float* __restrict__ ci,
                            float* __restrict__ a, float* __restrict__ b,
                            float* __restrict__ u, float* __restrict__ v) {
    __shared__ float sm[256];
    float sv = 0.f, si = 0.f;
    for (int j = threadIdx.x; j < Cc; j += blockDim.x) { sv += cv[j]; si += ci[j]; }
    sv = blockSum(sv, sm);
    si = blockSum(si, sm);
    float iv = 1.f / fmaxf(sv, 1.f), ii = 1.f / fmaxf(si, 1.f);
    for (int j = threadIdx.x; j < Cc; j += blockDim.x) {
        a[j] = cv[j] * iv; b[j] = ci[j] * ii; u[j] = 1.f; v[j] = 1.f;
    }
}

// ---------------- 8) Sinkhorn matvec (K is L2-resident, float4 rows) -------
__global__ void cma_matvec(const float* __restrict__ M, const float* __restrict__ x,
                           float* __restrict__ y) {
    __shared__ float sm[256];
    int r = blockIdx.x;
    const float4* Mr = (const float4*)(M + (size_t)r * Cc);
    const float4* x4 = (const float4*)x;
    float s = 0.f;
    for (int j = threadIdx.x; j < Cc / 4; j += blockDim.x) {
        float4 m = Mr[j], xx = x4[j];
        s += m.x * xx.x + m.y * xx.y + m.z * xx.z + m.w * xx.w;
    }
    s = blockSum(s, sm);
    if (threadIdx.x == 0) y[r] = s;
}

__global__ void cma_upd(const float* __restrict__ num, const float* __restrict__ den,
                        float* __restrict__ o) {
    int i = blockIdx.x * blockDim.x + threadIdx.x;
    if (i < Cc) { float n = num[i]; o[i] = (n > 0.f) ? n / (den[i] + 1e-8f) : 0.f; }
}

// ---------------- 9) transport -------------------------------------------
__global__ void cma_transport(const float* __restrict__ K, const float* __restrict__ u,
                              const float* __restrict__ v, float* __restrict__ t) {
    size_t idx = (size_t)blockIdx.x * blockDim.x + threadIdx.x;
    if (idx >= C2) return;
    int i = (int)(idx / Cc), j = (int)(idx % Cc);
    t[idx] = u[i] * K[idx] * v[j];
}

// ---------------- 10) normalize rows in place, conf/argmax -----------------
__global__ void cma_norm_rows(float* __restrict__ M, float* __restrict__ conf,
                              int* __restrict__ assign) {
    __shared__ float sm[256];
    __shared__ float smv[256];
    __shared__ int   smi[256];
    int r = blockIdx.x;
    float* row = M + (size_t)r * Cc;
    float s = 0.f;
    for (int j = threadIdx.x; j < Cc; j += blockDim.x) s += row[j];
    s = blockSum(s, sm);
    float inv = 1.f / fmaxf(s, 1e-12f);
    float bm = -1.f; int bi = Cc;
    for (int j = threadIdx.x; j < Cc; j += blockDim.x) {
        float v = row[j] * inv;
        row[j] = v;
        if (v > bm || (v == bm && j < bi)) { bm = v; bi = j; }
    }
    float ov; int oi;
    blockArgMax(bm, bi, smv, smi, ov, oi);
    if (threadIdx.x == 0) { conf[r] = ov; assign[r] = oi; }
}

// ---------------- 11) per-class flags --------------------------------------
__global__ void cma_flags(const float* __restrict__ actv, const float* __restrict__ acti,
                          const float* __restrict__ rc, const int* __restrict__ ra,
                          const float* __restrict__ cc, const int* __restrict__ ca,
                          int* __restrict__ common_row, int* __restrict__ spec_row,
                          int* __restrict__ rem_row, int* __restrict__ rem_col) {
    int i = blockIdx.x * blockDim.x + threadIdx.x;
    if (i >= Cc) return;
    bool av = actv[i] > 0.f, ai = acti[i] > 0.f;
    bool mutual_r = (ca[ra[i]] == i);
    bool cr = av && (rc[i] >= HIGH_TH) && mutual_r;
    common_row[i] = cr ? 1 : 0;
    spec_row[i]   = (av && (rc[i] >= HIGH_TH) && !cr) ? 1 : 0;
    rem_row[i]    = (av && (rc[i] >= MID_TH) && (rc[i] < HIGH_TH)) ? 1 : 0;
    rem_col[i]    = (ai && (cc[i] >= MID_TH) && (cc[i] < HIGH_TH)) ? 1 : 0;
}

// ---------------- 12) common / specific masks ------------------------------
__global__ void cma_masks(const float* __restrict__ row_t, const float* __restrict__ rc,
                          const int* __restrict__ ra, const int* __restrict__ common_row,
                          const int* __restrict__ spec_row,
                          float* __restrict__ common_rm, float* __restrict__ specific_rm) {
    size_t idx = (size_t)blockIdx.x * blockDim.x + threadIdx.x;
    if (idx >= C2) return;
    int i = (int)(idx / Cc), j = (int)(idx % Cc);
    common_rm[idx]   = (common_row[i] && ra[i] == j) ? rc[i] : 0.f;
    specific_rm[idx] = spec_row[i] ? row_t[idx] : 0.f;
}

// ---------------- 13) per-row top-3 mask -----------------------------------
__global__ void cma_topk(const float* __restrict__ M, const int* __restrict__ keep,
                         float* __restrict__ out) {
    __shared__ float smv[256];
    __shared__ int   smi[256];
    __shared__ int   seli[3];
    __shared__ float selv[3];
    int r = blockIdx.x;
    const float* row = M + (size_t)r * Cc;
    float* orow = out + (size_t)r * Cc;
    if (!keep[r]) {
        for (int j = threadIdx.x; j < Cc; j += blockDim.x) orow[j] = 0.f;
        return;
    }
    for (int round = 0; round < 3; round++) {
        float bm = -1e30f; int bi = Cc;
        for (int j = threadIdx.x; j < Cc; j += blockDim.x) {
            bool skip = false;
            for (int p = 0; p < round; p++) if (j == seli[p]) skip = true;
            if (skip) continue;
            float v = row[j];
            if (v > bm || (v == bm && j < bi)) { bm = v; bi = j; }
        }
        float ov; int oi;
        blockArgMax(bm, bi, smv, smi, ov, oi);
        if (threadIdx.x == 0) { seli[round] = oi; selv[round] = ov; }
        __syncthreads();
    }
    for (int j = threadIdx.x; j < Cc; j += blockDim.x) {
        float o = 0.f;
        if (j == seli[0]) o = selv[0];
        else if (j == seli[1]) o = selv[1];
        else if (j == seli[2]) o = selv[2];
        orow[j] = o;
    }
}

// ============================ orchestration ================================
extern "C" void kernel_launch(void* const* d_in, const int* in_sizes, int n_in,
                              void* d_out, int out_size, void* d_ws, size_t ws_size,
                              hipStream_t stream) {
    const float* vis_logits = (const float*)d_in[0];
    const float* ir_logits  = (const float*)d_in[1];
    const float* vis_memory = (const float*)d_in[2];
    const float* ir_memory  = (const float*)d_in[3];
    const int*   rgb_ids    = (const int*)d_in[4];
    const int*   ir_ids     = (const int*)d_in[5];

    float* out = (float*)d_out;            // 6 x [C,C]
    float* ws  = (float*)d_ws;

    // d_out slot aliasing: protos / proto_i^T / feat live in slots 2..5,
    // which are only written (as masks) after those tensors are dead.
    float* proto_iT = out + 2 * C2;
    float* proto_v  = out + 3 * C2;
    float* proto_i  = out + 4 * C2;
    float* feat     = out + 5 * C2;

    float* Kmat  = ws;                     // C2 floats
    float* KTmat = ws + C2;                // C2 floats
    unsigned short* vbf = (unsigned short*)(ws + 2 * C2);   // C*D ushorts
    unsigned short* ibf = vbf + (size_t)Cc * Dd;            // C*D ushorts
    float* vec = ws + 2 * C2 + (size_t)Cc * Dd;             // small vectors
    float* counts_v = vec + 0 * 2048;
    float* counts_i = vec + 1 * 2048;
    float* ent_v    = vec + 2 * 2048;
    float* ent_i    = vec + 3 * 2048;
    float* act_v    = vec + 4 * 2048;
    float* act_i    = vec + 5 * 2048;
    float* avec     = vec + 6 * 2048;
    float* bvec     = vec + 7 * 2048;
    float* uvec     = vec + 8 * 2048;
    float* vvec     = vec + 9 * 2048;
    float* kv       = vec + 10 * 2048;
    float* ktu      = vec + 11 * 2048;
    float* rowconf  = vec + 12 * 2048;
    float* colconf  = vec + 13 * 2048;
    int* rowassign  = (int*)(vec + 14 * 2048);
    int* colassign  = (int*)(vec + 15 * 2048);
    int* common_row = (int*)(vec + 16 * 2048);
    int* spec_row   = (int*)(vec + 17 * 2048);
    int* rem_row    = (int*)(vec + 18 * 2048);
    int* rem_col    = (int*)(vec + 19 * 2048);

    const int EW_GRID = (int)((C2 + 255) / 256);   // elementwise over C*C
    const int C_GRID  = (Cc + 255) / 256;
    const dim3 TR_GRID((Cc + 31) / 32, (Cc + 31) / 32);

    // re-init accumulators every call (graph-replay safe)
    hipMemsetAsync(proto_v, 0, 2 * C2 * sizeof(float), stream);
    hipMemsetAsync(counts_v, 0, 2 * 2048 * sizeof(float), stream);

    // softmax + class stats
    cma_sm_stats<<<NS, 256, 0, stream>>>(vis_logits, rgb_ids, proto_v, counts_v);
    cma_sm_stats<<<NS, 256, 0, stream>>>(ir_logits, ir_ids, proto_i, counts_i);
    cma_proto_fin<<<Cc, 256, 0, stream>>>(proto_v, counts_v, ent_v, act_v);
    cma_proto_fin<<<Cc, 256, 0, stream>>>(proto_i, counts_i, ent_i, act_i);

    // normalized bf16 memories + WMMA feature-similarity GEMM
    cma_norm_bf16<<<Cc, 256, 0, stream>>>(vis_memory, vbf);
    cma_norm_bf16<<<Cc, 256, 0, stream>>>(ir_memory, ibf);
    cma_feat_wmma<<<dim3(25, 25), 160, 0, stream>>>(vbf, ibf, feat);

    // coalesced ir_proto^T, then cost -> Sinkhorn kernel matrix + K^T
    cma_transpose<<<TR_GRID, 256, 0, stream>>>(proto_i, proto_iT);
    cma_cost_exp<<<EW_GRID, 256, 0, stream>>>(proto_v, proto_iT, feat, ent_v, ent_i,
                                              act_v, act_i, Kmat);
    cma_transpose<<<TR_GRID, 256, 0, stream>>>(Kmat, KTmat);
    cma_prep_ab<<<1, 256, 0, stream>>>(counts_v, counts_i, avec, bvec, uvec, vvec);

    // Sinkhorn iterations (K / K^T are L2-resident, 16 MB each)
    for (int it = 0; it < OT_ITERS; it++) {
        cma_matvec<<<Cc, 256, 0, stream>>>(Kmat, vvec, kv);
        cma_upd<<<C_GRID, 256, 0, stream>>>(avec, kv, uvec);
        cma_matvec<<<Cc, 256, 0, stream>>>(KTmat, uvec, ktu);
        cma_upd<<<C_GRID, 256, 0, stream>>>(bvec, ktu, vvec);
    }

    // transport (slot0), its transpose (slot1), then normalize in place
    cma_transport<<<EW_GRID, 256, 0, stream>>>(Kmat, uvec, vvec, out + 0 * C2);
    cma_transpose<<<TR_GRID, 256, 0, stream>>>(out + 0 * C2, out + 1 * C2);
    cma_norm_rows<<<Cc, 256, 0, stream>>>(out + 0 * C2, rowconf, rowassign);
    cma_norm_rows<<<Cc, 256, 0, stream>>>(out + 1 * C2, colconf, colassign);

    // flags + masks + top-3 masks
    cma_flags<<<C_GRID, 256, 0, stream>>>(act_v, act_i, rowconf, rowassign,
                                          colconf, colassign,
                                          common_row, spec_row, rem_row, rem_col);
    cma_masks<<<EW_GRID, 256, 0, stream>>>(out + 0 * C2, rowconf, rowassign,
                                           common_row, spec_row,
                                           out + 2 * C2, out + 3 * C2);
    cma_topk<<<Cc, 256, 0, stream>>>(out + 0 * C2, rem_row, out + 4 * C2);
    cma_topk<<<Cc, 256, 0, stream>>>(out + 1 * C2, rem_col, out + 5 * C2);
}
